// RecurrentGCN_28853590295019
// MI455X (gfx1250) — matrix operand order repaired and checked by reading, not compiled, not run on hardware
//
#include <hip/hip_runtime.h>
#include <hip/hip_bf16.h>

// ---------------------------------------------------------------------------
// RecurrentGCN forward on gfx1250 (MI455X). wave32, WMMA bf16 w/ fp32 accum.
// Exploits: h0==0  =>  R-gate dead, dconv uses only first-H weight rows.
// GEMM: 128x64 block / workgroup, B panel staged in LDS (+ TDM prefetch),
// one A fragment reused across 4 N-tiles per wave.
// ---------------------------------------------------------------------------

#define NN     6144      // nodes
#define FF     128       // input features
#define HH     256       // hidden
#define EE     (NN * 32) // edges
#define LLAY   2
#define NHEADS 4
#define HDIM   64

typedef __attribute__((ext_vector_type(16))) __bf16 v16bf;
typedef __attribute__((ext_vector_type(8)))  float  v8f;

#if defined(__has_builtin)
#  if __has_builtin(__builtin_amdgcn_tensor_load_to_lds) && __has_builtin(__builtin_amdgcn_s_wait_tensorcnt)
#    define HAVE_TDM 1
#  endif
#endif

#ifdef HAVE_TDM
typedef unsigned int u32x4 __attribute__((ext_vector_type(4)));
typedef int          i32x4 __attribute__((ext_vector_type(4)));
typedef int          i32x8 __attribute__((ext_vector_type(8)));
#endif

// float -> bf16, round-to-nearest-even (bit trick; avoids relying on native cast)
__device__ __forceinline__ __bf16 f2bf(float f) {
  unsigned u = __float_as_uint(f);
  unsigned r = (u + 0x7FFFu + ((u >> 16) & 1u)) >> 16;
  unsigned short hs = (unsigned short)r;
  return *reinterpret_cast<__bf16*>(&hs);
}

// ---------------------------------------------------------------------------
// Elementwise / prep kernels
// ---------------------------------------------------------------------------
__global__ void cast_kernel(const float* __restrict__ src, __bf16* __restrict__ dst, int n) {
  int i = blockIdx.x * blockDim.x + threadIdx.x;
  if (i < n) dst[i] = f2bf(src[i]);
}

// src f32 [NN,256] -> dst bf16 at column offset col_off with leading dim ld
__global__ void cast_cols_kernel(const float* __restrict__ src, __bf16* __restrict__ dst,
                                 int col_off, int ld, int count) {
  int idx = blockIdx.x * blockDim.x + threadIdx.x;
  if (idx >= count) return;
  int r = idx >> 8, c = idx & 255;
  dst[(size_t)r * ld + col_off + c] = f2bf(src[idx]);
}

// W [L,2,2,512,256] f32 -> Wcat [L,768,256] bf16 using only first-256 rows
// rows 0..255: W[l,0,0]+W[l,1,0]; 256..511: W[l,0,1]; 512..767: W[l,1,1]
__global__ void prep_dconv_kernel(const float* __restrict__ W, __bf16* __restrict__ dst) {
  int idx = blockIdx.x * blockDim.x + threadIdx.x;
  if (idx >= LLAY * 768 * 256) return;
  int l = idx / (768 * 256);
  int rem = idx % (768 * 256);
  int k = rem >> 8, n = rem & 255;
  const size_t blk = 512 * 256;
  float v;
  if (k < 256)
    v = W[(size_t)((l*2+0)*2+0) * blk + (size_t)k * 256 + n] +
        W[(size_t)((l*2+1)*2+0) * blk + (size_t)k * 256 + n];
  else if (k < 512)
    v = W[(size_t)((l*2+0)*2+1) * blk + (size_t)(k-256) * 256 + n];
  else
    v = W[(size_t)((l*2+1)*2+1) * blk + (size_t)(k-512) * 256 + n];
  dst[idx] = f2bf(v);
}

// dst[c*R + r] = src[r*C + c]   (f32 -> bf16)
__global__ void transpose_cast_kernel(const float* __restrict__ src, __bf16* __restrict__ dst,
                                      int R, int C) {
  int idx = blockIdx.x * blockDim.x + threadIdx.x;
  if (idx >= R * C) return;
  int r = idx / C, c = idx % C;
  dst[(size_t)c * R + r] = f2bf(src[idx]);
}

// ---------------------------------------------------------------------------
// Graph kernels
// ---------------------------------------------------------------------------
__global__ void deg_kernel(const int* __restrict__ row, const int* __restrict__ col,
                           float* __restrict__ dout, float* __restrict__ din) {
  int e = blockIdx.x * blockDim.x + threadIdx.x;
  if (e >= EE) return;
  atomicAdd(&dout[row[e]], 1.f);
  atomicAdd(&din[col[e]], 1.f);
}

// one wave per edge: agg_o[col] += (1/deg_out[row]) * h[row];  agg_i[col] += (1/deg_in[col]) * h[row]
__global__ void edge_agg_kernel(const int* __restrict__ row, const int* __restrict__ col,
                                const float* __restrict__ dout, const float* __restrict__ din,
                                const float* __restrict__ h,
                                float* __restrict__ agg_o, float* __restrict__ agg_i) {
  int e = (blockIdx.x * blockDim.x + threadIdx.x) >> 5;
  int lane = threadIdx.x & 31;
  if (e >= EE) return;
  int r = row[e], c = col[e];
  float no = 1.f / dout[r];
  float ni = 1.f / din[c];
  const float* hr = h + (size_t)r * HH;
  float* ao = agg_o + (size_t)c * HH;
  float* ai = agg_i + (size_t)c * HH;
  #pragma unroll
  for (int i = lane; i < HH; i += 32) {
    float v = hr[i];
    atomicAdd(ao + i, no * v);
    atomicAdd(ai + i, ni * v);
  }
}

// ---------------------------------------------------------------------------
// WMMA bf16 GEMM: C[M,N] = A[M,K] * B[K,N] (+bias, +relu).
// Block = 256 threads (8 waves) -> 128x64 tile. B panel (32x64) staged in LDS,
// shared by all 8 waves; each wave computes 16x64 with one A fragment reused
// across 4 WMMAs. Requires M%128==0, N%64==0, K%32==0 (true for all calls).
// mode: 0 = none, 1 = +bias, 2 = +bias then relu
// ---------------------------------------------------------------------------
#define GBM 128
#define GBN 64
#define GBK 32
#define BSTR 72   // padded LDS row stride (elements) to spread banks

__global__ void gemm_bf16_kernel(const __bf16* __restrict__ A, const __bf16* __restrict__ B,
                                 const float* __restrict__ bias, float* __restrict__ C,
                                 int M, int N, int K, int lda, int ldb, int ldc, int mode) {
  __shared__ __bf16 Bs[GBK * BSTR];
#ifdef HAVE_TDM
  __shared__ __bf16 Tdm[GBK * GBN];   // TDM prefetch landing zone (write-only)
#endif
  const int lane = threadIdx.x & 31;
  const int wave = threadIdx.x >> 5;
  const int nBlocks = N / GBN;
  const int bm = (blockIdx.x / nBlocks) * GBM;
  const int bn = (blockIdx.x % nBlocks) * GBN;
  const int lo = lane & 15, hi = lane >> 4;

#ifdef HAVE_TDM
  // Tensor Data Mover: async-prefetch the first 32x64 B panel of this block
  // into LDS scratch (warms L2/LDS path; compute does not depend on it).
  if (wave == 0) {
    unsigned lds_off = (unsigned)(size_t)(void*)Tdm;
    unsigned long long ga = (unsigned long long)(size_t)(const void*)(B + bn);
    u32x4 g0;
    g0[0] = 1u;                                        // count=1, user mode
    g0[1] = lds_off;                                   // lds_addr
    g0[2] = (unsigned)(ga & 0xFFFFFFFFu);              // global_addr[31:0]
    g0[3] = (unsigned)((ga >> 32) & 0x01FFFFFFu) | 0x80000000u; // addr[56:32] | type=2
    i32x8 g1;
    g1[0] = 0x00010000;                                // data_size=2B, no multicast
    g1[1] = (int)(((unsigned)N & 0xFFFFu) << 16);      // tensor_dim0[15:0]
    g1[2] = (int)((((unsigned)N >> 16) & 0xFFFFu) | (((unsigned)K & 0xFFFFu) << 16));
    g1[3] = (int)((((unsigned)K >> 16) & 0xFFFFu) | (64u << 16));  // tile_dim0=64
    g1[4] = 32;                                        // tile_dim1=32, tile_dim2=0
    g1[5] = (int)(unsigned)ldb;                        // tensor_dim0_stride[31:0]
    g1[6] = 0; g1[7] = 0;
    i32x4 g2 = {0, 0, 0, 0}, g3 = {0, 0, 0, 0};
#if defined(__clang_major__) && (__clang_major__ >= 23)
    i32x8 g4 = {0, 0, 0, 0, 0, 0, 0, 0};
    __builtin_amdgcn_tensor_load_to_lds(g0, g1, g2, g3, g4, 0);
#else
    __builtin_amdgcn_tensor_load_to_lds(g0, g1, g2, g3, 0);
#endif
    __builtin_amdgcn_s_wait_tensorcnt(0);
  }
#endif

  v8f acc[4];
  #pragma unroll
  for (int gI = 0; gI < 4; ++gI) acc[gI] = v8f{0.f, 0.f, 0.f, 0.f, 0.f, 0.f, 0.f, 0.f};

  // A fragment source: lane row = bm + wave*16 + lo; per-lane K base = hi*8
  const __bf16* ap = A + (size_t)(bm + wave * 16 + lo) * lda + hi * 8;

  // B panel staging: each of 256 threads copies 8 bf16 (16B) per k-step
  const int srow = threadIdx.x >> 3;          // 0..31
  const int scol = (threadIdx.x & 7) * 8;     // 0..56

  for (int k0 = 0; k0 < K; k0 += GBK) {
    const __bf16* bsrc = B + (size_t)(k0 + srow) * ldb + bn + scol;
    __bf16* bdst = Bs + srow * BSTR + scol;
    #pragma unroll
    for (int i = 0; i < 8; ++i) bdst[i] = bsrc[i];
    __syncthreads();

    v16bf a;
    #pragma unroll
    for (int i = 0; i < 8; ++i) { a[i] = ap[k0 + i]; a[8 + i] = ap[k0 + 16 + i]; }

    #pragma unroll
    for (int gI = 0; gI < 4; ++gI) {
      v16bf b;  // lane N column = gI*16+lo; elem i -> K = hi*8+(i&7)+(i>>3)*16
      const __bf16* bl = Bs + (hi * 8) * BSTR + gI * 16 + lo;
      #pragma unroll
      for (int i = 0; i < 16; ++i) b[i] = bl[((i & 7) + (i >> 3) * 16) * BSTR];
      acc[gI] = __builtin_amdgcn_wmma_f32_16x16x32_bf16(false, a, false, b, (short)0,
                                                        acc[gI], false, false);
    }
    __syncthreads();
  }

  #pragma unroll
  for (int gI = 0; gI < 4; ++gI) {
    const int cn = bn + gI * 16 + lo;
    #pragma unroll
    for (int r = 0; r < 8; ++r) {
      float v = acc[gI][r];
      if (mode >= 1) v += bias[cn];
      if (mode == 2) v = fmaxf(v, 0.f);
      C[(size_t)(bm + wave * 16 + r + hi * 8) * ldc + cn] = v;
    }
  }
}

// ---------------------------------------------------------------------------
// Row LayerNorm (256 cols, one wave per row) with fused pre-ops.
// mode 0: t = A                  mode 1: t = A + 0.5*relu((1-sigmoid(B))*tanh(Cc))
// mode 2: t = A + B
// relu_out applies relu after LN. outf (f32) and outbf (bf16, leading dim bf_ld) optional.
// ---------------------------------------------------------------------------
__global__ void row_ln_kernel(const float* __restrict__ A, const float* __restrict__ B,
                              const float* __restrict__ Cc, const float* __restrict__ g,
                              const float* __restrict__ be, float* __restrict__ outf,
                              __bf16* __restrict__ outbf, int bf_ld, int mode, int relu_out) {
  int wid = (blockIdx.x * blockDim.x + threadIdx.x) >> 5;
  int lane = threadIdx.x & 31;
  if (wid >= NN) return;
  size_t base = (size_t)wid * HH;
  float t[8];
  #pragma unroll
  for (int j = 0; j < 8; ++j) {
    size_t idx = base + lane + j * 32;
    float v = A[idx];
    if (mode == 1) {
      float z = 1.f / (1.f + __expf(-B[idx]));
      float hn = fmaxf((1.f - z) * tanhf(Cc[idx]), 0.f);
      v += 0.5f * hn;
    } else if (mode == 2) {
      v += B[idx];
    }
    t[j] = v;
  }
  float s = 0.f, ss = 0.f;
  #pragma unroll
  for (int j = 0; j < 8; ++j) { s += t[j]; ss += t[j] * t[j]; }
  #pragma unroll
  for (int off = 1; off < 32; off <<= 1) {
    s += __shfl_xor(s, off, 32);
    ss += __shfl_xor(ss, off, 32);
  }
  float mean = s * (1.f / 256.f);
  float var = ss * (1.f / 256.f) - mean * mean;
  float inv = rsqrtf(var + 1e-5f);
  #pragma unroll
  for (int j = 0; j < 8; ++j) {
    int c = lane + j * 32;
    float v = (t[j] - mean) * inv * g[c] + be[c];
    if (relu_out) v = fmaxf(v, 0.f);
    if (outf) outf[base + c] = v;
    if (outbf) outbf[(size_t)wid * bf_ld + c] = f2bf(v);
  }
}

// ---------------------------------------------------------------------------
// QKV split: qkv f32 [NN,768] (+bias) -> Q bf16 [head][n][d], K^T bf16 [head][d][n],
//            V bf16 [head][n][d]
// ---------------------------------------------------------------------------
__global__ void qkv_prep_kernel(const float* __restrict__ qkv, const float* __restrict__ bqkv,
                                __bf16* __restrict__ Qb, __bf16* __restrict__ Kt,
                                __bf16* __restrict__ Vb) {
  int idx = blockIdx.x * blockDim.x + threadIdx.x;
  if (idx >= NN * 768) return;
  int n = idx / 768, c = idx % 768;
  float v = qkv[idx] + bqkv[c];
  int which = c >> 8;
  int hh = (c & 255) >> 6;
  int d = c & 63;
  size_t hb = (size_t)hh * NN * HDIM;
  if (which == 0)      Qb[hb + (size_t)n * HDIM + d] = f2bf(v);
  else if (which == 1) Kt[hb + (size_t)d * NN + n]   = f2bf(v);
  else                 Vb[hb + (size_t)n * HDIM + d] = f2bf(v);
}

// ---------------------------------------------------------------------------
// Flash attention: one wave per (head, 16-row query tile). Online softmax over
// key blocks of 32. Scores & PV both via v_wmma_f32_16x16x32_bf16; the P tile
// is staged through LDS to convert C-layout f32 -> A-layout bf16.
// ---------------------------------------------------------------------------
__global__ void flash_attn_kernel(const __bf16* __restrict__ Qb, const __bf16* __restrict__ Kt,
                                  const __bf16* __restrict__ Vb, __bf16* __restrict__ Ob) {
  const int lane = threadIdx.x & 31;
  const int w = threadIdx.x >> 5;
  const int qt = blockIdx.x * 4 + w;   // 0..383
  const int hh = blockIdx.y;           // head
  const int lo = lane & 15, hi = lane >> 4;
  __shared__ float plds[4 * 16 * 32];
  float* P = plds + w * (16 * 32);

  const __bf16* Qh = Qb + (size_t)hh * NN * HDIM;
  const __bf16* Kh = Kt + (size_t)hh * NN * HDIM;  // [HDIM][NN]
  const __bf16* Vh = Vb + (size_t)hh * NN * HDIM;

  // Q fragments for K-dim chunks [0,32) and [32,64)
  v16bf aQ[2];
  const __bf16* qp = Qh + (size_t)(qt * 16 + lo) * HDIM;
  #pragma unroll
  for (int c = 0; c < 2; ++c) {
    int d0 = c * 32 + hi * 8;
    #pragma unroll
    for (int i = 0; i < 8; ++i) { aQ[c][i] = qp[d0 + i]; aQ[c][8 + i] = qp[d0 + 16 + i]; }
  }

  float rowm[8], rowl[8];
  #pragma unroll
  for (int r = 0; r < 8; ++r) { rowm[r] = -1e30f; rowl[r] = 0.f; }
  v8f accO[4];
  #pragma unroll
  for (int gI = 0; gI < 4; ++gI) accO[gI] = v8f{0.f, 0.f, 0.f, 0.f, 0.f, 0.f, 0.f, 0.f};

  for (int kb = 0; kb < NN; kb += 32) {
    // ---- scores for two 16-key tiles: S = Q (16x64) @ K^T (64x16) ----
    v8f sc[2];
    #pragma unroll
    for (int t = 0; t < 2; ++t) {
      sc[t] = v8f{0.f, 0.f, 0.f, 0.f, 0.f, 0.f, 0.f, 0.f};
      #pragma unroll
      for (int c = 0; c < 2; ++c) {
        v16bf bK;
        const __bf16* kp = Kh + (size_t)(c * 32 + hi * 8) * NN + kb + t * 16 + lo;
        #pragma unroll
        for (int i = 0; i < 16; ++i) bK[i] = kp[(size_t)((i & 7) + (i >> 3) * 16) * NN];
        sc[t] = __builtin_amdgcn_wmma_f32_16x16x32_bf16(false, aQ[c], false, bK, (short)0,
                                                        sc[t], false, false);
      }
    }
    // ---- online softmax (rows live across 16-lane halves) ----
    #pragma unroll
    for (int r = 0; r < 8; ++r) {
      float sa = sc[0][r] * 0.125f;  // 1/sqrt(64)
      float sb = sc[1][r] * 0.125f;
      float mx = fmaxf(sa, sb);
      #pragma unroll
      for (int off = 1; off < 16; off <<= 1) mx = fmaxf(mx, __shfl_xor(mx, off, 32));
      float mnew = fmaxf(rowm[r], mx);
      float alpha = __expf(rowm[r] - mnew);
      rowm[r] = mnew;
      float pa = __expf(sa - mnew), pb = __expf(sb - mnew);
      float rs = pa + pb;
      #pragma unroll
      for (int off = 1; off < 16; off <<= 1) rs += __shfl_xor(rs, off, 32);
      rowl[r] = rowl[r] * alpha + rs;
      #pragma unroll
      for (int gI = 0; gI < 4; ++gI) accO[gI][r] *= alpha;
      int rr = r + hi * 8;
      P[rr * 32 + lo] = pa;
      P[rr * 32 + 16 + lo] = pb;
    }
    asm volatile("s_wait_dscnt 0" ::: "memory");
    // ---- P (16x32) as A fragment, bf16 ----
    v16bf aP;
    {
      const float* pp = P + lo * 32 + hi * 8;
      #pragma unroll
      for (int i = 0; i < 8; ++i) { aP[i] = f2bf(pp[i]); aP[8 + i] = f2bf(pp[16 + i]); }
    }
    // ---- O += P @ V (V block: 32 keys x 64 dims, 4 N-chunks) ----
    #pragma unroll
    for (int gI = 0; gI < 4; ++gI) {
      v16bf bV;
      const __bf16* vp = Vh + (size_t)(kb + hi * 8) * HDIM + gI * 16 + lo;
      #pragma unroll
      for (int i = 0; i < 16; ++i) bV[i] = vp[((i & 7) + (i >> 3) * 16) * HDIM];
      accO[gI] = __builtin_amdgcn_wmma_f32_16x16x32_bf16(false, aP, false, bV, (short)0,
                                                         accO[gI], false, false);
    }
  }
  // ---- normalize & store: o[n][hh*64 + d], bf16 ----
  #pragma unroll
  for (int r = 0; r < 8; ++r) {
    float invl = 1.f / rowl[r];
    int rown = qt * 16 + r + hi * 8;
    #pragma unroll
    for (int gI = 0; gI < 4; ++gI)
      Ob[(size_t)rown * HH + hh * HDIM + gI * 16 + lo] = f2bf(accO[gI][r] * invl);
  }
}

// ---------------------------------------------------------------------------
// Final GEMV: out[n] = t2[n,:] . W3 + b3   (t2 already relu'd, f32 [NN,128])
// ---------------------------------------------------------------------------
__global__ void final_gemv_kernel(const float* __restrict__ t2, const float* __restrict__ W3,
                                  const float* __restrict__ b3, float* __restrict__ out) {
  int wid = (blockIdx.x * blockDim.x + threadIdx.x) >> 5;
  int lane = threadIdx.x & 31;
  if (wid >= NN) return;
  float s = 0.f;
  #pragma unroll
  for (int i = lane; i < 128; i += 32) s += t2[(size_t)wid * 128 + i] * W3[i];
  #pragma unroll
  for (int off = 1; off < 32; off <<= 1) s += __shfl_xor(s, off, 32);
  if (lane == 0) out[wid] = s + b3[0];
}

// ---------------------------------------------------------------------------
// Host launcher
// ---------------------------------------------------------------------------
extern "C" void kernel_launch(void* const* d_in, const int* in_sizes, int n_in,
                              void* d_out, int out_size, void* d_ws, size_t ws_size,
                              hipStream_t stream) {
  (void)in_sizes; (void)n_in; (void)out_size; (void)ws_size;
  const float* x       = (const float*)d_in[0];
  const int*   ei      = (const int*)d_in[1];
  const float* W_in    = (const float*)d_in[2];
  const float* b_in    = (const float*)d_in[3];
  const float* ln_in_g = (const float*)d_in[4];
  const float* ln_in_b = (const float*)d_in[5];
  const float* W_skip  = (const float*)d_in[6];
  const float* b_skip  = (const float*)d_in[7];
  const float* Wz      = (const float*)d_in[8];
  const float* bz      = (const float*)d_in[9];
  // d_in[10] (Wr), d_in[11] (br) are dead: R only multiplies the zero hidden state.
  const float* Wh      = (const float*)d_in[12];
  const float* bh      = (const float*)d_in[13];
  const float* ln_g    = (const float*)d_in[14];
  const float* ln_b    = (const float*)d_in[15];
  const float* Wqkv    = (const float*)d_in[16];
  const float* bqkv    = (const float*)d_in[17];
  const float* Wo      = (const float*)d_in[18];
  const float* bo      = (const float*)d_in[19];
  const float* ln_a_g  = (const float*)d_in[20];
  const float* ln_a_b  = (const float*)d_in[21];
  const float* W1      = (const float*)d_in[22];
  const float* b1      = (const float*)d_in[23];
  const float* ln_o_g  = (const float*)d_in[24];
  const float* ln_o_b  = (const float*)d_in[25];
  const float* W2      = (const float*)d_in[26];
  const float* b2      = (const float*)d_in[27];
  const float* W3      = (const float*)d_in[28];
  const float* b3      = (const float*)d_in[29];
  float* out = (float*)d_out;

  // ---- workspace carve-up ----
  char* p = (char*)d_ws;
  auto alloc = [&](size_t bytes) -> char* {
    char* r = p; p += (bytes + 255) & ~(size_t)255; return r;
  };
  float*  deg     = (float*)alloc((size_t)2 * NN * 4);
  float*  deg_out = deg;
  float*  deg_in  = deg + NN;
  __bf16* xb      = (__bf16*)alloc((size_t)NN * FF * 2);
  __bf16* hcat    = (__bf16*)alloc((size_t)NN * 768 * 2);   // [h | agg_o | agg_i] / [h | skip]
  float*  h       = (float*)alloc((size_t)NN * HH * 4);
  float*  skip    = (float*)alloc((size_t)NN * HH * 4);
  float*  agg     = (float*)alloc((size_t)2 * NN * HH * 4);
  float*  agg_o   = agg;
  float*  agg_i   = agg + (size_t)NN * HH;
  float*  tmpA    = (float*)alloc((size_t)NN * 768 * 4);    // qkv f32; also Zp/Hp
  float*  Zp      = tmpA;
  float*  Hp      = tmpA + (size_t)NN * HH;
  float*  tmpB    = (float*)alloc((size_t)NN * HH * 4);
  __bf16* Qb      = (__bf16*)alloc((size_t)NN * HH * 2);
  __bf16* Ktb     = (__bf16*)alloc((size_t)NN * HH * 2);
  __bf16* Vb      = (__bf16*)alloc((size_t)NN * HH * 2);
  __bf16* Obf     = (__bf16*)alloc((size_t)NN * HH * 2);
  __bf16* zbf     = (__bf16*)alloc((size_t)NN * HH * 2);
  float*  t2      = (float*)alloc((size_t)NN * 128 * 4);
  __bf16* W_in_b  = (__bf16*)alloc(128 * 256 * 2);
  __bf16* W_skip_b= (__bf16*)alloc(128 * 256 * 2);
  __bf16* WcatZ   = (__bf16*)alloc((size_t)LLAY * 768 * 256 * 2);
  __bf16* WcatH   = (__bf16*)alloc((size_t)LLAY * 768 * 256 * 2);
  __bf16* WqkvT_b = (__bf16*)alloc(256 * 768 * 2);
  __bf16* WoT_b   = (__bf16*)alloc(256 * 256 * 2);
  __bf16* W1_b    = (__bf16*)alloc(512 * 256 * 2);
  __bf16* W2_b    = (__bf16*)alloc(256 * 128 * 2);

  const int T = 256;
  auto g1 = [&](int n) { return (n + T - 1) / T; };
  // GEMM grid: (M/128) * (N/64) blocks of 256 threads
  auto ggemm = [&](int M, int N) { return (M / GBM) * (N / GBN); };

  // ---- degrees ----
  hipMemsetAsync(deg, 0, (size_t)2 * NN * 4, stream);
  deg_kernel<<<g1(EE), T, 0, stream>>>(ei, ei + EE, deg_out, deg_in);

  // ---- weight/input prep (bf16) ----
  cast_kernel<<<g1(128 * 256), T, 0, stream>>>(W_in, W_in_b, 128 * 256);
  cast_kernel<<<g1(128 * 256), T, 0, stream>>>(W_skip, W_skip_b, 128 * 256);
  cast_kernel<<<g1(512 * 256), T, 0, stream>>>(W1, W1_b, 512 * 256);
  cast_kernel<<<g1(256 * 128), T, 0, stream>>>(W2, W2_b, 256 * 128);
  cast_kernel<<<g1(NN * FF), T, 0, stream>>>(x, xb, NN * FF);
  prep_dconv_kernel<<<g1(LLAY * 768 * 256), T, 0, stream>>>(Wz, WcatZ);
  prep_dconv_kernel<<<g1(LLAY * 768 * 256), T, 0, stream>>>(Wh, WcatH);
  transpose_cast_kernel<<<g1(768 * 256), T, 0, stream>>>(Wqkv, WqkvT_b, 768, 256);
  transpose_cast_kernel<<<g1(256 * 256), T, 0, stream>>>(Wo, WoT_b, 256, 256);

  // ---- input projection + skip ----
  gemm_bf16_kernel<<<ggemm(NN, 256), T, 0, stream>>>(xb, W_in_b, b_in, tmpB,
                                                     NN, 256, 128, 128, 256, 256, 1);
  row_ln_kernel<<<768, T, 0, stream>>>(tmpB, nullptr, nullptr, ln_in_g, ln_in_b,
                                       h, hcat, 768, 0, 1);
  gemm_bf16_kernel<<<ggemm(NN, 256), T, 0, stream>>>(xb, W_skip_b, b_skip, skip,
                                                     NN, 256, 128, 128, 256, 256, 1);

  // ---- DCRNN layers ----
  for (int l = 0; l < LLAY; ++l) {
    hipMemsetAsync(agg, 0, (size_t)2 * NN * HH * 4, stream);
    edge_agg_kernel<<<EE / 8, T, 0, stream>>>(ei, ei + EE, deg_out, deg_in, h, agg_o, agg_i);
    cast_cols_kernel<<<g1(NN * HH), T, 0, stream>>>(agg_o, hcat, 256, 768, NN * HH);
    cast_cols_kernel<<<g1(NN * HH), T, 0, stream>>>(agg_i, hcat, 512, 768, NN * HH);
    gemm_bf16_kernel<<<ggemm(NN, 256), T, 0, stream>>>(hcat, WcatZ + (size_t)l * 768 * 256,
                                                       bz + l * 256, Zp,
                                                       NN, 256, 768, 768, 256, 256, 1);
    gemm_bf16_kernel<<<ggemm(NN, 256), T, 0, stream>>>(hcat, WcatH + (size_t)l * 768 * 256,
                                                       bh + l * 256, Hp,
                                                       NN, 256, 768, 768, 256, 256, 1);
    row_ln_kernel<<<768, T, 0, stream>>>(h, Zp, Hp, ln_g + l * 256, ln_b + l * 256,
                                         h, hcat, 768, 1, 0);
  }

  // ---- attention ----
  gemm_bf16_kernel<<<ggemm(NN, 768), T, 0, stream>>>(hcat, WqkvT_b, nullptr, tmpA,
                                                     NN, 768, 256, 768, 768, 768, 0);
  qkv_prep_kernel<<<g1(NN * 768), T, 0, stream>>>(tmpA, bqkv, Qb, Ktb, Vb);
  dim3 fgrid(NN / 16 / 4, NHEADS);
  flash_attn_kernel<<<fgrid, 128, 0, stream>>>(Qb, Ktb, Vb, Obf);
  gemm_bf16_kernel<<<ggemm(NN, 256), T, 0, stream>>>(Obf, WoT_b, bo, tmpB,
                                                     NN, 256, 256, 256, 256, 256, 1);
  row_ln_kernel<<<768, T, 0, stream>>>(h, tmpB, nullptr, ln_a_g, ln_a_b, h, hcat, 768, 2, 0);

  // ---- output MLP ----
  cast_cols_kernel<<<g1(NN * HH), T, 0, stream>>>(skip, hcat, 256, 768, NN * HH); // hcat=[h|skip]
  gemm_bf16_kernel<<<ggemm(NN, 256), T, 0, stream>>>(hcat, W1_b, b1, tmpB,
                                                     NN, 256, 512, 768, 256, 256, 1);
  row_ln_kernel<<<768, T, 0, stream>>>(tmpB, nullptr, nullptr, ln_o_g, ln_o_b,
                                       nullptr, zbf, 256, 0, 1);
  gemm_bf16_kernel<<<ggemm(NN, 128), T, 0, stream>>>(zbf, W2_b, b2, t2,
                                                     NN, 128, 256, 256, 128, 128, 2);
  final_gemv_kernel<<<768, T, 0, stream>>>(t2, W3, b3, out);
}
